// GraphHeadAttention_4157528343278
// MI455X (gfx1250) — matrix operand order repaired and checked by reading, not compiled, not run on hardware
//
#include <hip/hip_runtime.h>

// ---------------- types ----------------
typedef __bf16 bf16_t;
typedef __attribute__((ext_vector_type(8)))  __bf16 v8bf;
typedef __attribute__((ext_vector_type(16))) __bf16 v16bf;
typedef __attribute__((ext_vector_type(8)))  float  v8f;

#define NB 32      // batch / num nodes
#define NL 512     // seq len
#define ND 512     // model dim
#define NH 8       // heads
#define NHD 64     // head dim
#define NMAXD 8    // max path distance
#define NEDIM 128  // edge dim

// native f32 -> bf16 (RNE) conversion; lowers to v_cvt_pk_bf16_f32 on gfx1250
__device__ __forceinline__ bf16_t f2bf(float f) { return (bf16_t)f; }

__device__ __forceinline__ v16bf pack16(v8bf lo, v8bf hi) {
  v16bf r;
#pragma unroll
  for (int i = 0; i < 8; ++i) { r[i] = lo[i]; r[i + 8] = hi[i]; }
  return r;
}

__device__ __forceinline__ v8f wmma_bf16(v16bf a, v16bf b, v8f c) {
  return __builtin_amdgcn_wmma_f32_16x16x32_bf16(false, a, false, b, (short)0, c,
                                                 false, false);
}

__device__ __forceinline__ v8f vzero8() {
  v8f z = {0.f, 0.f, 0.f, 0.f, 0.f, 0.f, 0.f, 0.f};
  return z;
}

// ---------------- kernel 1: weight transpose + bf16 convert ----------------
// Wt[e][d] = bf16(W[d][e]), both 512x512
__global__ __launch_bounds__(256) void wprep_kernel(const float* __restrict__ W,
                                                    bf16_t* __restrict__ Wt) {
  int idx = blockIdx.x * 256 + threadIdx.x;  // 0 .. 512*512-1, coalesced read
  int d = idx >> 9;
  int e = idx & 511;
  Wt[(size_t)e * ND + d] = f2bf(W[idx]);
}

// ---------------- kernel 2: spatial + edge-encoding bias ----------------
// bias[b] = b_scale*mean_dst(b_vec[pl-1]) + c_scale*mean_dst(edge dot avg)
__global__ __launch_bounds__(1024) void bias_kernel(
    const float* __restrict__ edge_attr, const int* __restrict__ path_src,
    const int* __restrict__ path_edges, const int* __restrict__ path_lens,
    const float* __restrict__ edge_vector, const float* __restrict__ b_vec,
    const float* __restrict__ b_scale, const float* __restrict__ c_scale,
    float* __restrict__ bias_out) {
  __shared__ float acc[NB];
  int tid = threadIdx.x;
  if (tid < NB) acc[tid] = 0.f;
  __syncthreads();

  int p = tid;  // one thread per path (1024 paths)
  int pl = path_lens[p];
  if (pl > NMAXD) pl = NMAXD;
  float dsum = 0.f;
  for (int k = 0; k < pl; ++k) {
    const float* ev = edge_vector + k * NEDIM;
    const float* ea = edge_attr + (size_t)path_edges[p * NMAXD + k] * NEDIM;
    float s = 0.f;
#pragma unroll 4
    for (int e = 0; e < NEDIM; ++e) s += ev[e] * ea[e];
    dsum += s;
  }
  float cval = (pl > 0) ? dsum / (float)pl : 0.f;
  float bval = (pl > 0) ? b_vec[pl - 1] : 0.f;
  float contrib = (b_scale[0] * bval + c_scale[0] * cval) * (1.0f / (float)NB);
  atomicAdd(&acc[path_src[p]], contrib);
  __syncthreads();
  if (tid < NB) bias_out[tid] = acc[tid];
}

// ---------------- kernel 3: projection GEMM (fp32 X * bf16 Wt) -------------
// Each wave: 32 rows x 64 cols (2 A-frags x 4 B-frags -> 8 WMMA / k-step).
// out[b][h][l][hd] (bf16) = (X[m] . W[:,e]) + bias[e]; m=b*L+l, e=h*64+hd
__global__ __launch_bounds__(256) void proj_gemm_kernel(
    const float* __restrict__ X, const bf16_t* __restrict__ Wt,
    const float* __restrict__ bias, bf16_t* __restrict__ out) {
  const int tid = threadIdx.x;
  const int wave = tid >> 5, lane = tid & 31;
  const int ln = lane & 15, hh = lane >> 4;
  const int m0 = blockIdx.x * 256 + wave * 32;
  const int n0 = blockIdx.y * 64;

  v8f acc[2][4] = {{vzero8(), vzero8(), vzero8(), vzero8()},
                   {vzero8(), vzero8(), vzero8(), vzero8()}};

  for (int k0 = 0; k0 < ND; k0 += 32) {
    v16bf a[2];
#pragma unroll
    for (int g = 0; g < 2; ++g) {
      // A fragment: row (m0+16g+ln), k chunks [k0+8hh, +8) and [k0+16+8hh, +8)
      const float* xp = X + (size_t)(m0 + 16 * g + ln) * ND + k0 + 8 * hh;
      float4 a0 = *(const float4*)(xp);
      float4 a1 = *(const float4*)(xp + 4);
      float4 a2 = *(const float4*)(xp + 16);
      float4 a3 = *(const float4*)(xp + 20);
      v16bf v;
      v[0] = f2bf(a0.x);  v[1] = f2bf(a0.y);  v[2] = f2bf(a0.z);  v[3] = f2bf(a0.w);
      v[4] = f2bf(a1.x);  v[5] = f2bf(a1.y);  v[6] = f2bf(a1.z);  v[7] = f2bf(a1.w);
      v[8] = f2bf(a2.x);  v[9] = f2bf(a2.y);  v[10] = f2bf(a2.z); v[11] = f2bf(a2.w);
      v[12] = f2bf(a3.x); v[13] = f2bf(a3.y); v[14] = f2bf(a3.z); v[15] = f2bf(a3.w);
      a[g] = v;
    }
#pragma unroll
    for (int t = 0; t < 4; ++t) {
      // B fragment: Wt row (n0+t*16+ln), k chunks [k0+8hh, +8) and [k0+16+8hh, +8)
      const bf16_t* wp = Wt + (size_t)(n0 + t * 16 + ln) * ND + k0 + 8 * hh;
      v16bf b = pack16(*(const v8bf*)(wp), *(const v8bf*)(wp + 16));
      acc[0][t] = wmma_bf16(a[0], b, acc[0][t]);
      acc[1][t] = wmma_bf16(a[1], b, acc[1][t]);
    }
  }
#pragma unroll
  for (int g = 0; g < 2; ++g) {
#pragma unroll
    for (int t = 0; t < 4; ++t) {
      int e = n0 + t * 16 + ln;
      float bb = bias[e];
      int hidx = e >> 6, hd = e & 63;
#pragma unroll
      for (int r = 0; r < 8; ++r) {
        int m = m0 + 16 * g + r + 8 * hh;
        int bidx = m >> 9, l = m & (NL - 1);
        out[(((size_t)bidx * NH + hidx) * NL + l) * NHD + hd] =
            f2bf(acc[g][t][r] + bb);
      }
    }
  }
}

// ---------------- kernel 4: flash attention per (b, h, 128-row q tile) -----
__global__ __launch_bounds__(256) void attn_kernel(
    const bf16_t* __restrict__ qws, const bf16_t* __restrict__ kws,
    const bf16_t* __restrict__ vws, const float* __restrict__ mask,
    const float* __restrict__ biasb, bf16_t* __restrict__ ctx) {
  __shared__ __align__(16) bf16_t Vt[NHD * 32];       // V tile transposed [hd][key]
  __shared__ __align__(16) bf16_t Pb[8 * 16 * 32];    // per-wave P tile [16][32]

  const int tid = threadIdx.x;
  const int wave = tid >> 5, lane = tid & 31;
  const int ln = lane & 15, hh = lane >> 4;
  const int bid = blockIdx.x;
  const int qt = bid & 3;            // q tile (128 rows each)
  const int h = (bid >> 2) & 7;
  const int b = bid >> 5;

  const size_t base = ((size_t)b * NH + h) * NL * NHD;
  const bf16_t* qb = qws + base;
  const bf16_t* kb = kws + base;
  const bf16_t* vb = vws + base;
  const float* mrow = mask + (size_t)b * NL;
  const float bias = biasb[b];

  // Q fragments for this wave's 16 rows, d=0..63 (two K-steps)
  const int qrow = qt * 128 + wave * 16 + ln;
  const bf16_t* qp = qb + (size_t)qrow * NHD;
  v16bf aq[2];
#pragma unroll
  for (int kk = 0; kk < 2; ++kk) {
    const bf16_t* p = qp + kk * 32 + 16 * hh;
    aq[kk] = pack16(*(const v8bf*)(p), *(const v8bf*)(p + 8));
  }

  v8f o[4] = {vzero8(), vzero8(), vzero8(), vzero8()};
  float rmax[8], rsum[8];
#pragma unroll
  for (int r = 0; r < 8; ++r) { rmax[r] = -INFINITY; rsum[r] = 0.f; }

  bf16_t* pw = Pb + wave * 512;

  for (int kb0 = 0; kb0 < NL; kb0 += 32) {
    __syncthreads();
    // stage V tile transposed: Vt[hd][key] for 32 keys
    for (int i = tid; i < 32 * NHD; i += 256) {
      int key = i >> 6, hd = i & 63;
      Vt[hd * 32 + key] = vb[(size_t)(kb0 + key) * NHD + hd];
    }
    __syncthreads();

    // ---- S = Q K^T for two 16-key tiles ----
    v8f s0 = vzero8(), s1 = vzero8();
#pragma unroll
    for (int kk = 0; kk < 2; ++kk) {
      const bf16_t* kp0 = kb + (size_t)(kb0 + ln) * NHD + kk * 32 + 16 * hh;
      v16bf bk0 = pack16(*(const v8bf*)(kp0), *(const v8bf*)(kp0 + 8));
      s0 = wmma_bf16(aq[kk], bk0, s0);
      const bf16_t* kp1 = kb + (size_t)(kb0 + 16 + ln) * NHD + kk * 32 + 16 * hh;
      v16bf bk1 = pack16(*(const v8bf*)(kp1), *(const v8bf*)(kp1 + 8));
      s1 = wmma_bf16(aq[kk], bk1, s1);
    }

    const float add0 = bias + (1.0f - mrow[kb0 + ln]) * (-1e9f);
    const float add1 = bias + (1.0f - mrow[kb0 + 16 + ln]) * (-1e9f);

    // ---- online softmax (rows r+8*hh live across 16-lane halves) ----
    float corr[8];
#pragma unroll
    for (int r = 0; r < 8; ++r) {
      float v0 = s0[r] * 0.125f + add0;
      float v1 = s1[r] * 0.125f + add1;
      float m = fmaxf(v0, v1);
      for (int off = 1; off < 16; off <<= 1) m = fmaxf(m, __shfl_xor(m, off, 32));
      float mn = fmaxf(rmax[r], m);
      float c = __expf(rmax[r] - mn);
      rmax[r] = mn;
      float p0 = __expf(v0 - mn);
      float p1 = __expf(v1 - mn);
      float rs = p0 + p1;
      for (int off = 1; off < 16; off <<= 1) rs += __shfl_xor(rs, off, 32);
      rsum[r] = rsum[r] * c + rs;
      corr[r] = c;
      pw[(r + 8 * hh) * 32 + ln] = f2bf(p0);
      pw[(r + 8 * hh) * 32 + 16 + ln] = f2bf(p1);
    }
#pragma unroll
    for (int t = 0; t < 4; ++t)
#pragma unroll
      for (int r = 0; r < 8; ++r) o[t][r] *= corr[r];

    // wave-local LDS RAW fence (per-wave buffer; no workgroup barrier needed)
    asm volatile("s_wait_dscnt 0x0" ::: "memory");

    // ---- O += P V : A fragment of P from per-wave LDS ----
    const bf16_t* pr = pw + ln * 32 + 16 * hh;
    v16bf ap = pack16(*(const v8bf*)(pr), *(const v8bf*)(pr + 8));
#pragma unroll
    for (int t = 0; t < 4; ++t) {
      const bf16_t* vp = Vt + (t * 16 + ln) * 32 + 16 * hh;
      v16bf bv = pack16(*(const v8bf*)(vp), *(const v8bf*)(vp + 8));
      o[t] = wmma_bf16(ap, bv, o[t]);
    }
  }

  // ---- epilogue: normalize, write ctx[b][l][h*64+hd] as bf16 ----
#pragma unroll
  for (int t = 0; t < 4; ++t) {
    int hd = h * NHD + t * 16 + ln;
#pragma unroll
    for (int r = 0; r < 8; ++r) {
      int l = qt * 128 + wave * 16 + r + 8 * hh;
      ctx[((size_t)b * NL + l) * ND + hd] = f2bf(o[t][r] / rsum[r]);
    }
  }
}

// ---------------- kernel 5: output GEMM (bf16 ctx * bf16 WtO -> fp32) ------
// Each wave: 32 rows x 64 cols (2 A-frags x 4 B-frags -> 8 WMMA / k-step).
__global__ __launch_bounds__(256) void out_gemm_kernel(
    const bf16_t* __restrict__ X, const bf16_t* __restrict__ Wt,
    const float* __restrict__ bias, float* __restrict__ out) {
  const int tid = threadIdx.x;
  const int wave = tid >> 5, lane = tid & 31;
  const int ln = lane & 15, hh = lane >> 4;
  const int m0 = blockIdx.x * 256 + wave * 32;
  const int n0 = blockIdx.y * 64;

  v8f acc[2][4] = {{vzero8(), vzero8(), vzero8(), vzero8()},
                   {vzero8(), vzero8(), vzero8(), vzero8()}};

  for (int k0 = 0; k0 < ND; k0 += 32) {
    v16bf a[2];
#pragma unroll
    for (int g = 0; g < 2; ++g) {
      const bf16_t* xp = X + (size_t)(m0 + 16 * g + ln) * ND + k0 + 16 * hh;
      a[g] = pack16(*(const v8bf*)(xp), *(const v8bf*)(xp + 8));
    }
#pragma unroll
    for (int t = 0; t < 4; ++t) {
      const bf16_t* wp = Wt + (size_t)(n0 + t * 16 + ln) * ND + k0 + 16 * hh;
      v16bf b = pack16(*(const v8bf*)(wp), *(const v8bf*)(wp + 8));
      acc[0][t] = wmma_bf16(a[0], b, acc[0][t]);
      acc[1][t] = wmma_bf16(a[1], b, acc[1][t]);
    }
  }
#pragma unroll
  for (int g = 0; g < 2; ++g) {
#pragma unroll
    for (int t = 0; t < 4; ++t) {
      int e = n0 + t * 16 + ln;
      float bb = bias[e];
#pragma unroll
      for (int r = 0; r < 8; ++r) {
        int m = m0 + 16 * g + r + 8 * hh;
        out[(size_t)m * ND + e] = acc[g][t][r] + bb;
      }
    }
  }
}

// ---------------- launcher ----------------
extern "C" void kernel_launch(void* const* d_in, const int* in_sizes, int n_in,
                              void* d_out, int out_size, void* d_ws, size_t ws_size,
                              hipStream_t stream) {
  (void)in_sizes; (void)n_in; (void)out_size; (void)ws_size;
  const float* query = (const float*)d_in[0];
  const float* key   = (const float*)d_in[1];
  const float* value = (const float*)d_in[2];
  const float* mask  = (const float*)d_in[3];
  const float* edge_attr = (const float*)d_in[4];
  const int* path_src   = (const int*)d_in[5];
  const int* path_edges = (const int*)d_in[7];
  const int* path_lens  = (const int*)d_in[8];
  const float* WQ = (const float*)d_in[9];  const float* bQ = (const float*)d_in[10];
  const float* WK = (const float*)d_in[11]; const float* bK = (const float*)d_in[12];
  const float* WV = (const float*)d_in[13]; const float* bV = (const float*)d_in[14];
  const float* WO = (const float*)d_in[15]; const float* bO = (const float*)d_in[16];
  const float* edge_vector = (const float*)d_in[17];
  const float* b_vec = (const float*)d_in[18];
  const float* b_scale = (const float*)d_in[19];
  const float* c_scale = (const float*)d_in[20];
  float* out = (float*)d_out;

  // workspace layout
  const size_t QKV = (size_t)NB * NH * NL * NHD;  // 8,388,608 bf16 each
  const size_t WSZ = (size_t)ND * ND;             // 262,144 bf16 each
  bf16_t* qws = (bf16_t*)d_ws;
  bf16_t* kws = qws + QKV;
  bf16_t* vws = kws + QKV;
  bf16_t* ctx = vws + QKV;
  bf16_t* wtq = ctx + QKV;
  bf16_t* wtk = wtq + WSZ;
  bf16_t* wtv = wtk + WSZ;
  bf16_t* wto = wtv + WSZ;
  float* biasb = (float*)(wto + WSZ);

  // 1) weights -> transposed bf16
  wprep_kernel<<<1024, 256, 0, stream>>>(WQ, wtq);
  wprep_kernel<<<1024, 256, 0, stream>>>(WK, wtk);
  wprep_kernel<<<1024, 256, 0, stream>>>(WV, wtv);
  wprep_kernel<<<1024, 256, 0, stream>>>(WO, wto);

  // 2) per-batch attention bias
  bias_kernel<<<1, 1024, 0, stream>>>(edge_attr, path_src, path_edges, path_lens,
                                      edge_vector, b_vec, b_scale, c_scale, biasb);

  // 3) Q/K/V projections (M=16384 rows in 256-row block tiles)
  dim3 g(64, 8);
  proj_gemm_kernel<<<g, 256, 0, stream>>>(query, wtq, bQ, qws);
  proj_gemm_kernel<<<g, 256, 0, stream>>>(key,   wtk, bK, kws);
  proj_gemm_kernel<<<g, 256, 0, stream>>>(value, wtv, bV, vws);

  // 4) flash attention: B*H*4 q-tiles
  attn_kernel<<<NB * NH * 4, 256, 0, stream>>>(qws, kws, vws, mask, biasb, ctx);

  // 5) output projection
  out_gemm_kernel<<<g, 256, 0, stream>>>(ctx, wto, bO, out);
}
